// CausalAttention_4475355922925
// MI455X (gfx1250) — matrix operand order repaired and checked by reading, not compiled, not run on hardware
//
#include <hip/hip_runtime.h>
#include <hip/hip_bf16.h>
#include <stddef.h>
#include <stdint.h>

// ---------------------------------------------------------------------------
// Causal single-head attention, B=4, S=2048, D=1024.
// All matmul phases on the CDNA5 f16 WMMA pipe (v_wmma_f32_16x16x32_f16,
// fp32 accumulation). Operand staging uses GLOBAL_LOAD_ASYNC_TO_LDS_B128
// (ASYNCcnt) where no dtype conversion is needed, with double-buffered LDS
// and a single workgroup barrier per K-step.
//
// Workspace layout (80 MB total; resident in the 192 MB L2):
//   [ 0,16MB)  Qh  f16  (B*S*D)  -- pre-scaled by 1/sqrt(D)
//   [16,32MB)  Kh  f16
//   [32,48MB)  Vh  f16
//   [48,80MB)  Sb  f16  (B*S*S)  -- scores, overwritten in place by softmax P
// ---------------------------------------------------------------------------

typedef __attribute__((ext_vector_type(16))) _Float16 v16h;
typedef __attribute__((ext_vector_type(8)))  _Float16 v8h;
typedef __attribute__((ext_vector_type(8)))  float    v8f;
typedef __attribute__((ext_vector_type(4)))  float    v4f;
typedef __attribute__((ext_vector_type(4)))  int      v4i;

#define GAS __attribute__((address_space(1)))
#define LAS __attribute__((address_space(3)))

#if __has_builtin(__builtin_amdgcn_global_load_async_to_lds_b128)
#define HAVE_ASYNC_LDS 1
#else
#define HAVE_ASYNC_LDS 0
#endif

namespace {

constexpr int BATCH = 4;
constexpr int SEQ   = 2048;
constexpr int DIM   = 1024;
constexpr int BM    = 128;   // block tile M
constexpr int BN    = 128;   // block tile N
constexpr int BK    = 32;    // block tile K  (== WMMA K)
constexpr int LDSS  = 40;    // LDS row stride in halves (80B, multiple of 16B)

// 16-byte global -> LDS copy. Async (ASYNCcnt-tracked, no VGPR round trip)
// when the gfx1250 builtin is available; synchronous fallback otherwise.
// ROCm builtin signature (from diagnostics): (global int4*, lds int4*, Ii, Ii).
__device__ __forceinline__ void copy16_g2l(void* lds, const void* g) {
#if HAVE_ASYNC_LDS
  __builtin_amdgcn_global_load_async_to_lds_b128(
      (GAS v4i*)(uintptr_t)g,
      (LAS v4i*)(uint32_t)(uintptr_t)lds,  // low 32 bits == LDS byte offset
      /*imm_offset=*/0, /*cpol=*/0);
#else
  *(v8h*)lds = *(const v8h*)g;
#endif
}

__device__ __forceinline__ void wait_async0() {
#if HAVE_ASYNC_LDS
#if __has_builtin(__builtin_amdgcn_s_wait_asynccnt)
  __builtin_amdgcn_s_wait_asynccnt(0);
#else
  asm volatile("s_wait_asynccnt 0x0" ::: "memory");
#endif
#endif
}

__device__ __forceinline__ v8f wmma_f16(v16h a, v16h b, v8f c) {
  // D = A(16x32 f16) * B(32x16 f16) + C(16x16 f32)
  return __builtin_amdgcn_wmma_f32_16x16x32_f16(
      /*neg_a=*/false, a, /*neg_b=*/false, b,
      /*c_mod=*/(short)0, c, /*reuse_a=*/false, /*reuse_b=*/false);
}

__device__ __forceinline__ v16h make_frag(const _Float16* p0, const _Float16* p1) {
  v8h lo = *(const v8h*)p0;
  v8h hi = *(const v8h*)p1;
  return __builtin_shufflevector(lo, hi, 0, 1, 2, 3, 4, 5, 6, 7,
                                         8, 9, 10, 11, 12, 13, 14, 15);
}

// A-fragment (16x32 f16), LDS tile row-major [row][k], stride LDSS.
// row = lane&15; lanes 0-15 hold K 0..7/16..23, lanes 16-31 hold K 8..15/24..31.
__device__ __forceinline__ v16h load_a_frag(const _Float16* base, int row0, int lane) {
  const _Float16* p = base + (size_t)(row0 + (lane & 15)) * LDSS + 8 * (lane >> 4);
  return make_frag(p, p + 16);
}

// B-fragment (32x16 f16), LDS tile stored [col][k], stride LDSS.
// col = lane&15; lanes 0-15 hold K 0..7/16..23, lanes 16-31 hold K 8..15/24..31.
__device__ __forceinline__ v16h load_b_frag(const _Float16* base, int col0, int lane) {
  const _Float16* p = base + (size_t)(col0 + (lane & 15)) * LDSS + 8 * (lane >> 4);
  return make_frag(p, p + 16);
}

__device__ __forceinline__ v8f zero8() {
  v8f z = {0.f, 0.f, 0.f, 0.f, 0.f, 0.f, 0.f, 0.f};
  return z;
}

// ---------------------------------------------------------------------------
// Kernel 1: Y = X @ W (X fp32 8192x1024, W fp32 1024x1024) -> f16, 3 weights.
// Synchronous staging (fp32 -> f16 conversion in flight), double-buffered.
// ---------------------------------------------------------------------------
__global__ __launch_bounds__(256) void qkv_gemm_kernel(
    const float* __restrict__ x,
    const float* __restrict__ Wq, const float* __restrict__ Wk,
    const float* __restrict__ Wv,
    _Float16* __restrict__ Qh, _Float16* __restrict__ Kh,
    _Float16* __restrict__ Vh) {
  __shared__ __align__(16) _Float16 As[2][BM * LDSS];  // [m][k]
  __shared__ __align__(16) _Float16 Bs[2][BN * LDSS];  // [n][k] (W transposed)

  const int m0 = blockIdx.x * BM;   // over B*S
  const int n0 = blockIdx.y * BN;   // over D
  const int which = blockIdx.z;
  const float* W = (which == 0) ? Wq : (which == 1) ? Wk : Wv;
  _Float16* Y = (which == 0) ? Qh : (which == 1) ? Kh : Vh;
  const float outScale = (which == 0) ? 0.03125f : 1.0f;  // 1/sqrt(1024) into Q

  const int tid = threadIdx.x;
  const int lane = tid & 31, wave = tid >> 5;
  const int wm = wave & 3, wn = wave >> 2;   // 4x2 wave grid -> 32x64 per wave

  auto stage = [&](int buf, int k0) {
    {  // X tile 128x32 fp32 -> f16
      const int r = tid >> 3;            // 0..31
      const int cc = (tid & 7) * 4;      // 0..28
      for (int it = 0; it < 4; ++it) {
        const int row = r + it * 32;
        v4f f = *(const v4f*)(x + (size_t)(m0 + row) * DIM + k0 + cc);
        _Float16* d = &As[buf][(size_t)row * LDSS + cc];
        d[0] = (_Float16)f[0]; d[1] = (_Float16)f[1];
        d[2] = (_Float16)f[2]; d[3] = (_Float16)f[3];
      }
    }
    {  // W tile 32x128 fp32 transposed -> Bs[n][k] f16
      const int kk = tid >> 5;           // 0..7
      const int nn = (tid & 31) * 4;     // 0..124
      for (int it = 0; it < 4; ++it) {
        const int krow = kk + it * 8;
        v4f f = *(const v4f*)(W + (size_t)(k0 + krow) * DIM + n0 + nn);
        Bs[buf][(size_t)(nn + 0) * LDSS + krow] = (_Float16)f[0];
        Bs[buf][(size_t)(nn + 1) * LDSS + krow] = (_Float16)f[1];
        Bs[buf][(size_t)(nn + 2) * LDSS + krow] = (_Float16)f[2];
        Bs[buf][(size_t)(nn + 3) * LDSS + krow] = (_Float16)f[3];
      }
    }
  };

  v8f acc[2][4];
  for (int i = 0; i < 2; ++i)
    for (int j = 0; j < 4; ++j) acc[i][j] = zero8();

  constexpr int KITERS = DIM / BK;
  stage(0, 0);
  for (int kk = 0; kk < KITERS; ++kk) {
    __syncthreads();  // flushes prior DS stores; protects both buffers
    if (kk + 1 < KITERS) stage((kk + 1) & 1, (kk + 1) * BK);
    const _Float16* Ab = As[kk & 1];
    const _Float16* Bb = Bs[kk & 1];
    v16h a[2], b[4];
    for (int i = 0; i < 2; ++i) a[i] = load_a_frag(Ab, wm * 32 + i * 16, lane);
    for (int j = 0; j < 4; ++j) b[j] = load_b_frag(Bb, wn * 64 + j * 16, lane);
    for (int i = 0; i < 2; ++i)
      for (int j = 0; j < 4; ++j) acc[i][j] = wmma_f16(a[i], b[j], acc[i][j]);
  }

  // Epilogue: C layout -> M = v + 8*(lane>=16), N = lane&15.
  const int h = lane >> 4, c = lane & 15;
  for (int i = 0; i < 2; ++i)
    for (int j = 0; j < 4; ++j) {
      const int row0 = m0 + wm * 32 + i * 16 + 8 * h;
      const int col = n0 + wn * 64 + j * 16 + c;
      for (int v = 0; v < 8; ++v)
        Y[(size_t)(row0 + v) * DIM + col] = (_Float16)(acc[i][j][v] * outScale);
    }
}

// ---------------------------------------------------------------------------
// Kernel 2: S = Q @ K^T, lower-triangular 128x128 blocks only (block-causal).
// K is row-major [s][d] so K^T B-fragments are contiguous over d.
// Q/K tiles staged with async global->LDS, double-buffered.
// ---------------------------------------------------------------------------
__global__ __launch_bounds__(256) void scores_kernel(
    const _Float16* __restrict__ Qh, const _Float16* __restrict__ Kh,
    _Float16* __restrict__ Sb) {
  const int qb = blockIdx.x, kb = blockIdx.y, b = blockIdx.z;
  if (kb > qb) return;

  __shared__ __align__(16) _Float16 Qs[2][BM * LDSS];  // [q][d]
  __shared__ __align__(16) _Float16 Ks[2][BM * LDSS];  // [k][d]

  const _Float16* Qb = Qh + (size_t)b * SEQ * DIM;
  const _Float16* Kb = Kh + (size_t)b * SEQ * DIM;
  const int q0 = qb * BM, k0g = kb * BN;

  const int tid = threadIdx.x;
  const int lane = tid & 31, wave = tid >> 5;
  const int wm = wave & 3, wn = wave >> 2;

  const int r = tid >> 2;            // 0..63
  const int ch = (tid & 3) * 8;      // 0,8,16,24
  auto stage = [&](int buf, int d0) {
    for (int it = 0; it < 2; ++it) {
      const int row = r + it * 64;
      copy16_g2l(&Qs[buf][(size_t)row * LDSS + ch],
                 Qb + (size_t)(q0 + row) * DIM + d0 + ch);
      copy16_g2l(&Ks[buf][(size_t)row * LDSS + ch],
                 Kb + (size_t)(k0g + row) * DIM + d0 + ch);
    }
  };

  v8f acc[2][4];
  for (int i = 0; i < 2; ++i)
    for (int j = 0; j < 4; ++j) acc[i][j] = zero8();

  constexpr int KITERS = DIM / BK;
  stage(0, 0);
  for (int kk = 0; kk < KITERS; ++kk) {
    wait_async0();     // own async copies for buf kk&1 complete
    __syncthreads();   // everyone's complete; prior reads of other buf done
    if (kk + 1 < KITERS) stage((kk + 1) & 1, (kk + 1) * BK);
    const _Float16* Qt = Qs[kk & 1];
    const _Float16* Kt = Ks[kk & 1];
    v16h a[2], bfr[4];
    for (int i = 0; i < 2; ++i) a[i] = load_a_frag(Qt, wm * 32 + i * 16, lane);
    for (int j = 0; j < 4; ++j) bfr[j] = load_b_frag(Kt, wn * 64 + j * 16, lane);
    for (int i = 0; i < 2; ++i)
      for (int j = 0; j < 4; ++j) acc[i][j] = wmma_f16(a[i], bfr[j], acc[i][j]);
  }

  _Float16* Srow = Sb + (size_t)b * SEQ * SEQ;
  const int h = lane >> 4, c = lane & 15;
  for (int i = 0; i < 2; ++i)
    for (int j = 0; j < 4; ++j) {
      const int row0 = q0 + wm * 32 + i * 16 + 8 * h;
      const int col = k0g + wn * 64 + j * 16 + c;
      for (int v = 0; v < 8; ++v)
        Srow[(size_t)(row0 + v) * SEQ + col] = (_Float16)acc[i][j][v];
    }
}

// ---------------------------------------------------------------------------
// Kernel 3: exact causal softmax, one wave32 per query row, in place.
// Writes P = exp(s-m)/l for k<=q and 0 for q<k<ceil128(q+1) so the PV GEMM
// can consume full block-aligned tiles.
// ---------------------------------------------------------------------------
__global__ __launch_bounds__(256) void softmax_kernel(_Float16* __restrict__ Sb) {
  const int lane = threadIdx.x & 31;
  const int wave = threadIdx.x >> 5;
  const long gidx = (long)blockIdx.x * 8 + wave;  // row in [0, B*S)
  const int b = (int)(gidx >> 11);                // / SEQ
  const int q = (int)(gidx & (SEQ - 1));
  _Float16* row = Sb + (size_t)b * SEQ * SEQ + (size_t)q * SEQ;
  const int n = q + 1;                        // valid keys
  const int nfill = ((q >> 7) + 1) << 7;      // zero-fill to 128-block edge

  float m = -1e30f;
  for (int k = lane; k < n; k += 32) m = fmaxf(m, (float)row[k]);
  for (int off = 16; off; off >>= 1) m = fmaxf(m, __shfl_xor(m, off));

  float ssum = 0.f;
  for (int k = lane; k < n; k += 32) ssum += __expf((float)row[k] - m);
  for (int off = 16; off; off >>= 1) ssum += __shfl_xor(ssum, off);
  const float inv = 1.0f / ssum;

  for (int k = lane; k < nfill; k += 32) {
    float p = (k < n) ? __expf((float)row[k] - m) * inv : 0.0f;
    row[k] = (_Float16)p;
  }
}

// ---------------------------------------------------------------------------
// Kernel 4: out = P @ V (fp32 out), k-loop truncated at causal block boundary.
// P staged async (contiguous f16); V staged synchronously with transpose so
// B-fragments are contiguous over k. Double-buffered, one barrier per step.
// ---------------------------------------------------------------------------
__global__ __launch_bounds__(256) void pv_kernel(
    const _Float16* __restrict__ Pm, const _Float16* __restrict__ Vh,
    float* __restrict__ out) {
  const int qb = blockIdx.x, nb = blockIdx.y, b = blockIdx.z;
  const int q0 = qb * BM, n0 = nb * BN;

  __shared__ __align__(16) _Float16 Ps[2][BM * LDSS];  // [q][k]
  __shared__ __align__(16) _Float16 Vs[2][BN * LDSS];  // [n][k] (V transposed)

  const _Float16* Pb = Pm + (size_t)b * SEQ * SEQ;
  const _Float16* Vb = Vh + (size_t)b * SEQ * DIM;

  const int tid = threadIdx.x;
  const int lane = tid & 31, wave = tid >> 5;
  const int wm = wave & 3, wn = wave >> 2;

  const int r = tid >> 2;            // 0..63
  const int ch = (tid & 3) * 8;      // 0,8,16,24
  auto stage_p = [&](int buf, int k0) {
    for (int it = 0; it < 2; ++it) {
      const int row = r + it * 64;
      copy16_g2l(&Ps[buf][(size_t)row * LDSS + ch],
                 Pb + (size_t)(q0 + row) * SEQ + k0 + ch);
    }
  };
  const int kk_ = tid >> 4;          // 0..15
  const int nn_ = (tid & 15) * 8;    // 0..120
  auto stage_v = [&](int buf, int k0) {
    for (int it = 0; it < 2; ++it) {
      const int krow = kk_ + it * 16;
      v8h f = *(const v8h*)(Vb + (size_t)(k0 + krow) * DIM + n0 + nn_);
      for (int j = 0; j < 8; ++j)
        Vs[buf][(size_t)(nn_ + j) * LDSS + krow] = f[j];
    }
  };

  v8f acc[2][4];
  for (int i = 0; i < 2; ++i)
    for (int j = 0; j < 4; ++j) acc[i][j] = zero8();

  const int kiters = (q0 + BM) / BK;  // causal: keys >= q0+128 never contribute
  stage_p(0, 0);
  stage_v(0, 0);
  for (int kk = 0; kk < kiters; ++kk) {
    wait_async0();
    __syncthreads();
    if (kk + 1 < kiters) {
      stage_p((kk + 1) & 1, (kk + 1) * BK);
      stage_v((kk + 1) & 1, (kk + 1) * BK);
    }
    const _Float16* Pt = Ps[kk & 1];
    const _Float16* Vt = Vs[kk & 1];
    v16h a[2], bfr[4];
    for (int i = 0; i < 2; ++i) a[i] = load_a_frag(Pt, wm * 32 + i * 16, lane);
    for (int j = 0; j < 4; ++j) bfr[j] = load_b_frag(Vt, wn * 64 + j * 16, lane);
    for (int i = 0; i < 2; ++i)
      for (int j = 0; j < 4; ++j) acc[i][j] = wmma_f16(a[i], bfr[j], acc[i][j]);
  }

  const int h = lane >> 4, c = lane & 15;
  float* Ob = out + (size_t)b * SEQ * DIM;
  for (int i = 0; i < 2; ++i)
    for (int j = 0; j < 4; ++j) {
      const int row0 = q0 + wm * 32 + i * 16 + 8 * h;
      const int col = n0 + wn * 64 + j * 16 + c;
      for (int v = 0; v < 8; ++v)
        Ob[(size_t)(row0 + v) * DIM + col] = acc[i][j][v];
    }
}

}  // namespace

extern "C" void kernel_launch(void* const* d_in, const int* in_sizes, int n_in,
                              void* d_out, int out_size, void* d_ws, size_t ws_size,
                              hipStream_t stream) {
  (void)in_sizes; (void)n_in; (void)out_size; (void)ws_size;
  const float* x  = (const float*)d_in[0];
  const float* Wq = (const float*)d_in[1];
  const float* Wk = (const float*)d_in[2];
  const float* Wv = (const float*)d_in[3];
  float* out = (float*)d_out;

  char* ws = (char*)d_ws;
  const size_t qkv_bytes = (size_t)BATCH * SEQ * DIM * sizeof(_Float16);  // 16 MB
  _Float16* Qh = (_Float16*)(ws);
  _Float16* Kh = (_Float16*)(ws + qkv_bytes);
  _Float16* Vh = (_Float16*)(ws + 2 * qkv_bytes);
  _Float16* Sb = (_Float16*)(ws + 3 * qkv_bytes);  // 32 MB scores/P

  const int M = BATCH * SEQ;  // 8192

  qkv_gemm_kernel<<<dim3(M / BM, DIM / BN, 3), 256, 0, stream>>>(
      x, Wq, Wk, Wv, Qh, Kh, Vh);
  scores_kernel<<<dim3(SEQ / BM, SEQ / BN, BATCH), 256, 0, stream>>>(Qh, Kh, Sb);
  softmax_kernel<<<dim3((size_t)BATCH * SEQ / 8), 256, 0, stream>>>(Sb);
  pv_kernel<<<dim3(SEQ / BM, DIM / BN, BATCH), 256, 0, stream>>>(Sb, Vh, out);
}
